// MultiHeadAttention_52527450030485
// MI455X (gfx1250) — matrix operand order repaired and checked by reading, not compiled, not run on hardware
//
#include <hip/hip_runtime.h>

// ---------------------------------------------------------------------------
// MHA for MI455X (gfx1250, wave32, WMMA).  B=2, S=2048, D=2048, H=16, Dh=128.
// GEMMs and flash-attention KV tiles: LDS-staged, double-buffered via
// GLOBAL_LOAD_ASYNC_TO_LDS_B128 (ASYNCcnt / s_wait_asynccnt); all matrix math
// via v_wmma_f32_16x16x32_f16 with fp32 accumulation.
// ---------------------------------------------------------------------------

typedef __attribute__((ext_vector_type(16))) _Float16 v16h;
typedef __attribute__((ext_vector_type(8)))  _Float16 v8h;
typedef __attribute__((ext_vector_type(8)))  float    v8f;

#define SEQ    2048
#define DMODEL 2048
#define NHEAD  16
#define HDIM   128
#define BATCH  2
#define MROWS  (BATCH * SEQ)       // 4096

__device__ inline v8f wmma16(v16h a, v16h b, v8f c) {
  return __builtin_amdgcn_wmma_f32_16x16x32_f16(
      /*neg_a=*/false, a, /*neg_b=*/false, b,
      /*c_mod=*/(short)0, c, /*reuse_a=*/false, /*reuse_b=*/false);
}

__device__ inline v8f vzero8() {
  v8f z = {0.f, 0.f, 0.f, 0.f, 0.f, 0.f, 0.f, 0.f};
  return z;
}

// ---- CDNA5 async global -> LDS copy (16B per lane), tracked by ASYNCcnt ----
__device__ inline void async_copy16(void* lds_dst, const void* gsrc) {
  unsigned loff = (unsigned)(unsigned long long)lds_dst;   // low 32b of generic = LDS offset
  unsigned long long ga = (unsigned long long)gsrc;
  asm volatile("global_load_async_to_lds_b128 %0, %1, off"
               :: "v"(loff), "v"(ga)
               : "memory");
}

__device__ inline void wait_async0() {
#if __has_builtin(__builtin_amdgcn_s_wait_asynccnt)
  __builtin_amdgcn_s_wait_asynccnt(0);
#else
  asm volatile("s_wait_asynccnt 0" ::: "memory");
#endif
}

// --- A-fragment (16x32 f16, MxK).  Lane l<16: row M=l, K=0..7 & 16..23;
//     lane l>=16: row M=l, K=8..15 & 24..31 (ISA 7.12.2).
__device__ inline v16h load_a_frag(const _Float16* __restrict__ A, int lda, int m0, int k0) {
  const int lane = threadIdx.x & 31;
  const _Float16* p = A + (size_t)(m0 + (lane & 15)) * lda + k0 + ((lane & 16) ? 8 : 0);
  v8h lo = *(const v8h*)p;
  v8h hi = *(const v8h*)(p + 16);
  v16h r;
#pragma unroll
  for (int i = 0; i < 8; ++i) { r[i] = lo[i]; r[8 + i] = hi[i]; }
  return r;
}

// --- B-fragment (32x16 f16, KxN) with B[k][n] = Mrow[n][k].
//     Lane l<16: col N=l, K=0..7 & 16..23; lane l>=16: col N=l, K=8..15 & 24..31.
__device__ inline v16h load_b_frag(const _Float16* __restrict__ W, int ldw, int n0, int k0) {
  const int lane = threadIdx.x & 31;
  const _Float16* p = W + (size_t)(n0 + (lane & 15)) * ldw + k0 + ((lane & 16) ? 8 : 0);
  v8h lo = *(const v8h*)p;
  v8h hi = *(const v8h*)(p + 16);
  v16h r;
#pragma unroll
  for (int i = 0; i < 8; ++i) { r[i] = lo[i]; r[8 + i] = hi[i]; }
  return r;
}

// ---------------------------------------------------------------------------
// Vectorized f32 -> f16 conversion (8 elems / thread).
// ---------------------------------------------------------------------------
__global__ __launch_bounds__(256) void cvt_f32_f16_kernel(const float* __restrict__ s,
                                                          _Float16* __restrict__ d, int n8) {
  int i = blockIdx.x * 256 + threadIdx.x;
  if (i >= n8) return;
  const float4* sp = (const float4*)s + (size_t)i * 2;
  float4 a = sp[0], b = sp[1];
  v8h o;
  o[0] = (_Float16)a.x; o[1] = (_Float16)a.y; o[2] = (_Float16)a.z; o[3] = (_Float16)a.w;
  o[4] = (_Float16)b.x; o[5] = (_Float16)b.y; o[6] = (_Float16)b.z; o[7] = (_Float16)b.w;
  *(v8h*)(d + (size_t)i * 8) = o;
}

// ---------------------------------------------------------------------------
// GEMM: out[M=4096][N=2048] = A[M][2048] * W[N][2048]^T   (both f16).
// Block = 256 threads = 8 waves; block tile 128x64; K-step 32.
// Tiles staged in LDS via async DMA, double buffered.
// mode 0: f16 row-major   mode 1: f16 head-transposed (Vt)   mode 2: f32.
// ---------------------------------------------------------------------------
__global__ __launch_bounds__(256) void gemm_wmma_kernel(const _Float16* __restrict__ A,
                                                        const _Float16* __restrict__ W,
                                                        void* __restrict__ outp,
                                                        int mode) {
  __shared__ _Float16 At[2][128 * 32];   // 8 KB per buffer
  __shared__ _Float16 Bt[2][64 * 32];    // 4 KB per buffer

  const int tid  = threadIdx.x;
  const int wave = tid >> 5;
  const int lane = tid & 31;
  const int m0 = blockIdx.x * 128;
  const int n0 = blockIdx.y * 64;

  const int crow = tid >> 2;             // 0..63  (one 16B chunk per thread)
  const int cch  = (tid & 3) * 8;        // half offset of 16B chunk within row

  v8f acc[4];
#pragma unroll
  for (int t = 0; t < 4; ++t) acc[t] = vzero8();

  {
    const int k0 = 0;
    async_copy16(&At[0][crow * 32 + cch],        A + (size_t)(m0 + crow) * DMODEL + k0 + cch);
    async_copy16(&At[0][(64 + crow) * 32 + cch], A + (size_t)(m0 + 64 + crow) * DMODEL + k0 + cch);
    async_copy16(&Bt[0][crow * 32 + cch],        W + (size_t)(n0 + crow) * DMODEL + k0 + cch);
  }

  for (int kt = 0; kt < DMODEL / 32; ++kt) {
    const int cur = kt & 1;
    wait_async0();           // this wave's DMA into buf[cur] done
    __syncthreads();         // everyone's DMA visible

    if (kt + 1 < DMODEL / 32) {
      const int k0 = (kt + 1) * 32;
      async_copy16(&At[cur ^ 1][crow * 32 + cch],        A + (size_t)(m0 + crow) * DMODEL + k0 + cch);
      async_copy16(&At[cur ^ 1][(64 + crow) * 32 + cch], A + (size_t)(m0 + 64 + crow) * DMODEL + k0 + cch);
      async_copy16(&Bt[cur ^ 1][crow * 32 + cch],        W + (size_t)(n0 + crow) * DMODEL + k0 + cch);
    }

    const _Float16* ap = &At[cur][(wave * 16 + (lane & 15)) * 32 + ((lane & 16) ? 8 : 0)];
    v8h lo = *(const v8h*)ap;
    v8h hi = *(const v8h*)(ap + 16);
    v16h a;
#pragma unroll
    for (int i = 0; i < 8; ++i) { a[i] = lo[i]; a[8 + i] = hi[i]; }

#pragma unroll
    for (int t = 0; t < 4; ++t) {
      v16h b = load_b_frag(&Bt[cur][0], 32, t * 16, 0);
      acc[t] = wmma16(a, b, acc[t]);
    }
    __syncthreads();         // all reads of buf[cur] done before refill
  }

  const int nloc = lane & 15;
  const int mo = (lane & 16) ? 8 : 0;
  if (mode == 0) {
    _Float16* O = (_Float16*)outp;
#pragma unroll
    for (int t = 0; t < 4; ++t)
#pragma unroll
      for (int r = 0; r < 8; ++r)
        O[(size_t)(m0 + wave * 16 + mo + r) * DMODEL + n0 + t * 16 + nloc] = (_Float16)acc[t][r];
  } else if (mode == 1) {
    _Float16* O = (_Float16*)outp;   // Vt[b][h][d][s]
#pragma unroll
    for (int t = 0; t < 4; ++t)
#pragma unroll
      for (int r = 0; r < 8; ++r) {
        int row = m0 + wave * 16 + mo + r;     // b*2048 + s
        int col = n0 + t * 16 + nloc;          // h*128 + d
        O[((size_t)((row >> 11) * DMODEL + col)) * SEQ + (row & 2047)] = (_Float16)acc[t][r];
      }
  } else {
    float* O = (float*)outp;
#pragma unroll
    for (int t = 0; t < 4; ++t)
#pragma unroll
      for (int r = 0; r < 8; ++r)
        O[(size_t)(m0 + wave * 16 + mo + r) * DMODEL + n0 + t * 16 + nloc] = acc[t][r];
  }
}

// ---------------------------------------------------------------------------
// RoPE in-place on q,k (f16, layout [b*s][2048]).
// ---------------------------------------------------------------------------
__global__ __launch_bounds__(256) void rope_kernel(_Float16* __restrict__ qh,
                                                   _Float16* __restrict__ kh) {
  int idx = blockIdx.x * blockDim.x + threadIdx.x;     // 0 .. B*S*1024
  if (idx >= BATCH * SEQ * (DMODEL / 2)) return;
  int t = idx & 1023;
  int row = idx >> 10;
  int s = row & 2047;
  int h = t >> 6;
  int tl = t & 63;
  int f = (2 * tl < 64) ? (2 * tl) : (2 * tl - 64);
  float ang = (float)s * __powf(10000.f, -(float)f * (1.0f / 64.0f));
  float sn, cs;
  __sincosf(ang, &sn, &cs);
  size_t off = (size_t)row * DMODEL + h * HDIM + 2 * tl;
  {
    float x1 = (float)qh[off], x2 = (float)qh[off + 1];
    qh[off]     = (_Float16)(x1 * cs - x2 * sn);
    qh[off + 1] = (_Float16)(x2 * cs + x1 * sn);
  }
  {
    float x1 = (float)kh[off], x2 = (float)kh[off + 1];
    kh[off]     = (_Float16)(x1 * cs - x2 * sn);
    kh[off + 1] = (_Float16)(x2 * cs + x1 * sn);
  }
}

// ---------------------------------------------------------------------------
// Flash attention, causal.  Block = 4 waves sharing (b,h); each wave owns a
// 16-row Q tile.  K/V tiles (32 keys) staged block-wide in LDS via async DMA,
// double buffered.  Uniform per-block KV trip count keeps barriers legal;
// waves past their causal range skip compute only.
// ---------------------------------------------------------------------------
__device__ inline void stage_kv(const _Float16* __restrict__ Kb, const _Float16* __restrict__ Vb,
                                _Float16* Kt, _Float16* Vt, int kv0, int tid) {
  // K tile: 32 rows x 128 halves (256B/row) -> 512 x 16B chunks
#pragma unroll
  for (int j = 0; j < 4; ++j) {
    int id = j * 128 + tid;
    int kr = id >> 4;              // 0..31
    int kc = (id & 15) * 8;        // halves
    async_copy16(&Kt[kr * 128 + kc], Kb + (size_t)(kv0 + kr) * DMODEL + kc);
  }
  // V tile (head-transposed): 128 d-rows x 32 keys (64B/row) -> 512 x 16B chunks
#pragma unroll
  for (int j = 0; j < 4; ++j) {
    int id = j * 128 + tid;
    int dr = id >> 2;              // 0..127
    int dc = (id & 3) * 8;         // halves
    async_copy16(&Vt[dr * 32 + dc], Vb + (size_t)dr * SEQ + kv0 + dc);
  }
}

__global__ __launch_bounds__(128) void flash_attn_kernel(const _Float16* __restrict__ Q,
                                                         const _Float16* __restrict__ K,
                                                         const _Float16* __restrict__ Vt,
                                                         _Float16* __restrict__ O) {
  __shared__ _Float16 Kt[2][32 * 128];     // 8 KB per buffer
  __shared__ _Float16 Vtl[2][128 * 32];    // 8 KB per buffer
  __shared__ _Float16 plds[4][16 * 32];    // per-wave P staging

  const int tid  = threadIdx.x;
  const int wave = tid >> 5;
  const int lane = tid & 31;
  const int bh = blockIdx.y;
  const int b = bh >> 4, h = bh & 15;
  const int q0 = (blockIdx.x * 4 + wave) * 16;

  const _Float16* Qb = Q + (size_t)b * SEQ * DMODEL + h * HDIM;
  const _Float16* Kb = K + (size_t)b * SEQ * DMODEL + h * HDIM;
  const _Float16* Vb = Vt + (size_t)(b * NHEAD + h) * HDIM * SEQ;

  v16h qa[4];
#pragma unroll
  for (int c = 0; c < 4; ++c) qa[c] = load_a_frag(Qb, DMODEL, q0, c * 32);

  v8f o[8];
#pragma unroll
  for (int t = 0; t < 8; ++t) o[t] = vzero8();
  float mi[8], li[8];
#pragma unroll
  for (int r = 0; r < 8; ++r) { mi[r] = -1e30f; li[r] = 0.f; }

  const int nloc = lane & 15;
  const int mo = (lane & 16) ? 8 : 0;
  const int nkv = 2 * (blockIdx.x + 1);    // uniform across block (causal max)

  stage_kv(Kb, Vb, Kt[0], Vtl[0], 0, tid);

  for (int it = 0; it < nkv; ++it) {
    const int kv0 = it * 32;
    const int cur = it & 1;
    wait_async0();
    __syncthreads();
    if (it + 1 < nkv) stage_kv(Kb, Vb, Kt[cur ^ 1], Vtl[cur ^ 1], kv0 + 32, tid);

    if (kv0 <= q0 + 15) {                  // this wave's causal range
      v8f s0 = vzero8(), s1 = vzero8();
#pragma unroll
      for (int c = 0; c < 4; ++c) {
        v16h b0 = load_b_frag(&Kt[cur][0], 128, 0,  c * 32);
        v16h b1 = load_b_frag(&Kt[cur][0], 128, 16, c * 32);
        s0 = wmma16(qa[c], b0, s0);
        s1 = wmma16(qa[c], b1, s1);
      }

      float p0[8], p1[8], al[8];
#pragma unroll
      for (int r = 0; r < 8; ++r) {
        const int rowq = q0 + r + mo;
        float v0 = s0[r] * 0.0883883476f;     // 1/sqrt(128)
        float v1 = s1[r] * 0.0883883476f;
        if (kv0 + nloc > rowq)      v0 = -1e30f;
        if (kv0 + 16 + nloc > rowq) v1 = -1e30f;
        float loc = fmaxf(v0, v1);
#pragma unroll
        for (int off = 1; off <= 8; off <<= 1) loc = fmaxf(loc, __shfl_xor(loc, off));
        float nm = fmaxf(mi[r], loc);
        al[r] = __expf(mi[r] - nm);
        float e0 = __expf(v0 - nm);
        float e1 = __expf(v1 - nm);
        float rs = e0 + e1;
#pragma unroll
        for (int off = 1; off <= 8; off <<= 1) rs += __shfl_xor(rs, off);
        li[r] = li[r] * al[r] + rs;
        mi[r] = nm;
        p0[r] = e0; p1[r] = e1;
      }
#pragma unroll
      for (int t = 0; t < 8; ++t)
#pragma unroll
        for (int r = 0; r < 8; ++r) o[t][r] *= al[r];

      // D-layout P -> row-major 16x32 in per-wave LDS scratch (same-wave DS order)
      _Float16* pw = &plds[wave][0];
#pragma unroll
      for (int r = 0; r < 8; ++r) {
        const int row = r + mo;
        pw[row * 32 + nloc]      = (_Float16)p0[r];
        pw[row * 32 + 16 + nloc] = (_Float16)p1[r];
      }
      v16h pa = load_a_frag((const _Float16*)pw, 32, 0, 0);

#pragma unroll
      for (int t = 0; t < 8; ++t) {
        v16h vb = load_b_frag(&Vtl[cur][0], 32, t * 16, 0);
        o[t] = wmma16(pa, vb, o[t]);
      }
    }
    __syncthreads();     // all reads of buf[cur] done before refill
  }

  _Float16* Ob = O + (size_t)b * SEQ * DMODEL + h * HDIM;
#pragma unroll
  for (int r = 0; r < 8; ++r) li[r] = 1.0f / li[r];
#pragma unroll
  for (int t = 0; t < 8; ++t)
#pragma unroll
    for (int r = 0; r < 8; ++r)
      Ob[(size_t)(q0 + r + mo) * DMODEL + t * 16 + nloc] = (_Float16)(o[t][r] * li[r]);
}

// ---------------------------------------------------------------------------
// Host launcher.  Workspace: qh|kh|vth|ah (4x16MB) + xh (16MB) + wh (8MB).
// ---------------------------------------------------------------------------
extern "C" void kernel_launch(void* const* d_in, const int* in_sizes, int n_in,
                              void* d_out, int out_size, void* d_ws, size_t ws_size,
                              hipStream_t stream) {
  const float* x  = (const float*)d_in[0];
  const float* Wq = (const float*)d_in[1];
  const float* Wk = (const float*)d_in[2];
  const float* Wv = (const float*)d_in[3];
  const float* Wo = (const float*)d_in[4];
  float* out = (float*)d_out;

  const size_t BUF = (size_t)MROWS * DMODEL * sizeof(_Float16);   // 16 MB
  char* ws = (char*)d_ws;
  _Float16* qh  = (_Float16*)(ws + 0 * BUF);
  _Float16* kh  = (_Float16*)(ws + 1 * BUF);
  _Float16* vth = (_Float16*)(ws + 2 * BUF);   // Vt[b][h][d][s]
  _Float16* ah  = (_Float16*)(ws + 3 * BUF);
  _Float16* xh  = (_Float16*)(ws + 4 * BUF);
  _Float16* wh  = (_Float16*)(ws + 5 * BUF);   // reused per weight matrix

  const int xN8 = MROWS * DMODEL / 8;
  const int wN8 = DMODEL * DMODEL / 8;

  dim3 ggrid(MROWS / 128, DMODEL / 64);
  dim3 gblk(256);

  cvt_f32_f16_kernel<<<(xN8 + 255) / 256, 256, 0, stream>>>(x, xh, xN8);

  cvt_f32_f16_kernel<<<(wN8 + 255) / 256, 256, 0, stream>>>(Wq, wh, wN8);
  gemm_wmma_kernel<<<ggrid, gblk, 0, stream>>>(xh, wh, (void*)qh, 0);

  cvt_f32_f16_kernel<<<(wN8 + 255) / 256, 256, 0, stream>>>(Wk, wh, wN8);
  gemm_wmma_kernel<<<ggrid, gblk, 0, stream>>>(xh, wh, (void*)kh, 0);

  cvt_f32_f16_kernel<<<(wN8 + 255) / 256, 256, 0, stream>>>(Wv, wh, wN8);
  gemm_wmma_kernel<<<ggrid, gblk, 0, stream>>>(xh, wh, (void*)vth, 1);

  int rope_threads = BATCH * SEQ * (DMODEL / 2);
  rope_kernel<<<(rope_threads + 255) / 256, 256, 0, stream>>>(qh, kh);

  flash_attn_kernel<<<dim3(SEQ / 64, BATCH * NHEAD), 128, 0, stream>>>(qh, kh, vth, ah);

  cvt_f32_f16_kernel<<<(wN8 + 255) / 256, 256, 0, stream>>>(Wo, wh, wN8);
  gemm_wmma_kernel<<<ggrid, gblk, 0, stream>>>(ah, wh, (void*)out, 2);
}
